// TransformerEncoder_28209345200422
// MI455X (gfx1250) — matrix operand order repaired and checked by reading, not compiled, not run on hardware
//
#include <hip/hip_runtime.h>
#include <hip/hip_bf16.h>

// ---------------------------------------------------------------------------
// Graph TransformerConv (PyG-style), 3 layers, N=50000, E=800000, D=256, H=4, C=64
// GEMMs via v_wmma_f32_16x16x32_bf16 (bf16 A/B, f32 accum), 16x64 per wave
// with A-fragment register reuse (4 WMMA per A load pair).
// Edge phase: wave-per-edge, L2-resident gathers + f32 atomics.
// ---------------------------------------------------------------------------

typedef __attribute__((ext_vector_type(16))) __bf16 v16bf;
typedef __attribute__((ext_vector_type(8)))  __bf16 v8bf;
typedef __attribute__((ext_vector_type(8)))  float  v8f;

#define DCH 256          // feature dim (H*C)
#define HEADS 4
#define CH 64
#define ATT_SCALE 0.125f // 1/sqrt(64)

// ---------------------------- conversions ----------------------------------

__global__ void convert_h_bf16(const float* __restrict__ hin,
                               __bf16* __restrict__ hbf, int n) {
    int i = blockIdx.x * blockDim.x + threadIdx.x;
    if (i < n) hbf[i] = (__bf16)hin[i];
}

// Convert + transpose the 4 weight matrices for this layer.
// W[k][n] (k-major) -> Wt[n][k] (n-major) so B fragments load contiguously.
__global__ void convert_w4(const float* __restrict__ Wq, const float* __restrict__ Wk,
                           const float* __restrict__ Wv, const float* __restrict__ Ws,
                           __bf16* __restrict__ Wt) {
    int idx = blockIdx.x * blockDim.x + threadIdx.x;   // idx = k*256 + n
    if (idx >= DCH * DCH) return;
    int k = idx >> 8;
    int n = idx & 255;
    int t = n * DCH + k;
    Wt[t]                 = (__bf16)Wq[idx];
    Wt[t +   DCH * DCH]   = (__bf16)Wk[idx];
    Wt[t + 2 * DCH * DCH] = (__bf16)Wv[idx];
    Wt[t + 3 * DCH * DCH] = (__bf16)Ws[idx];
}

// ------------------------------- GEMM --------------------------------------
// C[M x 256] = A[M x 256](bf16) * Bt^T + bias.
// One wave computes a 16x64 strip: one A fragment feeds 4 WMMAs per K-step
// (A register reuse -> 10 b128 loads per 4 wmma instead of 16).
// M is a multiple of 16; guard is wave-uniform so EXEC stays all-ones.
__global__ void gemm_bf16_wmma(const __bf16* __restrict__ A,   // [M][256]
                               const __bf16* __restrict__ Bt,  // [256][256], Bt[n][k]
                               const float*  __restrict__ bias,// [256]
                               float* __restrict__ Cout,       // [M][256]
                               int M) {
    int wave = (blockIdx.x * blockDim.x + threadIdx.x) >> 5;
    int lane = threadIdx.x & 31;
    const int colGroups = DCH / 64;                // 4 groups of 64 cols
    int tm = wave / colGroups;
    int tg = wave % colGroups;
    if (tm * 16 >= M) return;                      // uniform per wave

    int half = lane >> 4;                          // 0/1
    int l16  = lane & 15;
    const __bf16* Arow = A  + (size_t)(tm * 16 + l16) * DCH;
    const __bf16* Brow = Bt + (size_t)(tg * 64 + l16) * DCH;   // n-subtile 0

    v8f acc[4] = {};
#pragma unroll
    for (int k0 = 0; k0 < DCH; k0 += 32) {
        // A fragment (16x32 bf16): lane m=l16; elem i -> K=k0+half*16+i
        // (two aligned 16B loads)
        v16bf a;
        ((v8bf*)&a)[0] = *(const v8bf*)(Arow + k0 + half * 16);
        ((v8bf*)&a)[1] = *(const v8bf*)(Arow + k0 + half * 16 + 8);
#pragma unroll
        for (int j = 0; j < 4; ++j) {
            // B fragment (32x16 bf16): lane n=l16; elem i -> K=k0+half*16+i
            // (one aligned 32B load from transposed weights)
            v16bf b = *(const v16bf*)(Brow + (size_t)j * 16 * DCH + k0 + half * 16);
            acc[j] = __builtin_amdgcn_wmma_f32_16x16x32_bf16(
                false, a, false, b, (short)0, acc[j], false, false);
        }
    }

#pragma unroll
    for (int j = 0; j < 4; ++j) {
        int col = tg * 64 + j * 16 + l16;
        float bv = bias[col];
#pragma unroll
        for (int v = 0; v < 8; ++v) {
            int row = tm * 16 + v + 8 * half;      // C layout: VGPR v -> M=v+8*half
            Cout[(size_t)row * DCH + col] = acc[j][v] + bv;
        }
    }
}

// ---------------------------- edge phase -----------------------------------

__device__ __forceinline__ void atomicMaxF(float* addr, float val) {
    // int-max for non-negative, uint-min for negative; -inf init pattern works
    if (val >= 0.0f) atomicMax((int*)addr, __float_as_int(val));
    else             atomicMin((unsigned int*)addr, (unsigned int)__float_as_int(val));
}

__global__ void init_layer(float* __restrict__ amax, float* __restrict__ denom,
                           float* __restrict__ agg, int nNH, int nAgg) {
    int i = blockIdx.x * blockDim.x + threadIdx.x;
    if (i < nNH) { amax[i] = -INFINITY; denom[i] = 0.0f; }
    if (i < nAgg) agg[i] = 0.0f;
}

// One wave per edge: 8 lanes per head, each lane an 8-channel partial dot,
// xor-shuffle reduce within head, atomic max into segment max.
__global__ void edge_logits(const float* __restrict__ q, const float* __restrict__ k,
                            const int* __restrict__ src, const int* __restrict__ dst,
                            float* __restrict__ alpha, float* __restrict__ amax, int E) {
    int e    = (blockIdx.x * blockDim.x + threadIdx.x) >> 5;
    int lane = threadIdx.x & 31;
    if (e >= E) return;
    int s = src[e], d = dst[e];
    int head = lane >> 3;
    int ch   = head * CH + (lane & 7) * 8;
    const float* qp = q + (size_t)d * DCH + ch;
    const float* kp = k + (size_t)s * DCH + ch;
    float p = 0.0f;
#pragma unroll
    for (int j = 0; j < 8; ++j) p += qp[j] * kp[j];
    p += __shfl_xor(p, 1, 32);
    p += __shfl_xor(p, 2, 32);
    p += __shfl_xor(p, 4, 32);
    if ((lane & 7) == 0) {
        float a = p * ATT_SCALE;
        alpha[(size_t)e * HEADS + head] = a;
        atomicMaxF(&amax[(size_t)d * HEADS + head], a);
    }
}

// One thread per (edge, head): ex = exp(alpha - amax[dst]), accumulate denom.
__global__ void edge_exp(float* __restrict__ alpha, const int* __restrict__ dst,
                         const float* __restrict__ amax, float* __restrict__ denom,
                         int EH) {
    int i = blockIdx.x * blockDim.x + threadIdx.x;
    if (i >= EH) return;
    int e = i >> 2, h = i & 3;
    int d = dst[e];
    float ex = __expf(alpha[i] - amax[d * HEADS + h]);
    alpha[i] = ex;
    atomicAdd(&denom[d * HEADS + h], ex);
}

// One wave per edge: normalize and scatter-add v[src]*attn into agg[dst].
__global__ void edge_scatter(const float* __restrict__ ex, const float* __restrict__ denom,
                             const float* __restrict__ v,
                             const int* __restrict__ src, const int* __restrict__ dst,
                             float* __restrict__ agg, int E) {
    int e    = (blockIdx.x * blockDim.x + threadIdx.x) >> 5;
    int lane = threadIdx.x & 31;
    if (e >= E) return;
    int s = src[e], d = dst[e];
    int head = lane >> 3;
    float w = ex[(size_t)e * HEADS + head] /
              (denom[(size_t)d * HEADS + head] + 1e-16f);
    int ch = head * CH + (lane & 7) * 8;
    const float* vp = v   + (size_t)s * DCH + ch;
    float*       ap = agg + (size_t)d * DCH + ch;
#pragma unroll
    for (int j = 0; j < 8; ++j) atomicAdd(&ap[j], vp[j] * w);
}

// h = relu(agg + s), in place over the h/output buffer (s stored there).
__global__ void combine_relu(const float* __restrict__ agg, float* __restrict__ h, int n) {
    int i = blockIdx.x * blockDim.x + threadIdx.x;
    if (i < n) {
        float x = agg[i] + h[i];
        h[i] = x > 0.0f ? x : 0.0f;
    }
}

// ------------------------------ driver -------------------------------------

extern "C" void kernel_launch(void* const* d_in, const int* in_sizes, int n_in,
                              void* d_out, int out_size, void* d_ws, size_t ws_size,
                              hipStream_t stream) {
    const float* x  = (const float*)d_in[0];
    const int*   ei = (const int*)d_in[1];
    const float* Wq = (const float*)d_in[2];
    const float* bq = (const float*)d_in[3];
    const float* Wk = (const float*)d_in[4];
    const float* bk = (const float*)d_in[5];
    const float* Wv = (const float*)d_in[6];
    const float* bv = (const float*)d_in[7];
    const float* Ws = (const float*)d_in[8];
    const float* bs = (const float*)d_in[9];

    const int N = in_sizes[0] / DCH;     // 50000
    const int E = in_sizes[1] / 2;       // 800000
    const int* src = ei;
    const int* dst = ei + E;
    float* out = (float*)d_out;          // doubles as the h buffer

    // ------- workspace carve-up -------
    const size_t szNF = (size_t)N * DCH * sizeof(float);     // 51.2 MB
    char* w = (char*)d_ws;
    float*  qb    = (float*)w;            w += szNF;
    float*  kb    = (float*)w;            w += szNF;
    float*  vb    = (float*)w;            w += szNF;
    float*  agg   = (float*)w;            w += szNF;
    __bf16* hbf   = (__bf16*)w;           w += (size_t)N * DCH * sizeof(__bf16);
    __bf16* Wt    = (__bf16*)w;           w += (size_t)4 * DCH * DCH * sizeof(__bf16);
    float*  alpha = (float*)w;            w += (size_t)E * HEADS * sizeof(float);
    float*  amax  = (float*)w;            w += (size_t)N * HEADS * sizeof(float);
    float*  denom = (float*)w;            w += (size_t)N * HEADS * sizeof(float);
    (void)ws_size; (void)n_in; (void)out_size;

    const int NF  = N * DCH;                          // 12.8M
    const int tpb = 256;
    const int gridNF   = (NF + tpb - 1) / tpb;
    const int gridW    = (DCH * DCH + tpb - 1) / tpb;
    const int gemmWaves = (N / 16) * (DCH / 64);      // 16x64 strip per wave
    const int gridGemm = (gemmWaves + (tpb / 32) - 1) / (tpb / 32);
    const int gridEdge = (E * 32 + tpb - 1) / tpb;    // wave per edge
    const int gridEH   = (E * HEADS + tpb - 1) / tpb;

    for (int l = 0; l < 3; ++l) {
        const float* hin = (l == 0) ? x : out;
        const size_t wOff = (size_t)l * DCH * DCH;
        const size_t bOff = (size_t)l * DCH;

        convert_h_bf16<<<gridNF, tpb, 0, stream>>>(hin, hbf, NF);
        convert_w4<<<gridW, tpb, 0, stream>>>(Wq + wOff, Wk + wOff, Wv + wOff, Ws + wOff, Wt);

        gemm_bf16_wmma<<<gridGemm, tpb, 0, stream>>>(hbf, Wt,                 bq + bOff, qb,  N);
        gemm_bf16_wmma<<<gridGemm, tpb, 0, stream>>>(hbf, Wt +     DCH * DCH, bk + bOff, kb,  N);
        gemm_bf16_wmma<<<gridGemm, tpb, 0, stream>>>(hbf, Wt + 2 * DCH * DCH, bv + bOff, vb,  N);
        gemm_bf16_wmma<<<gridGemm, tpb, 0, stream>>>(hbf, Wt + 3 * DCH * DCH, bs + bOff, out, N); // skip term s -> h buffer

        init_layer<<<gridNF, tpb, 0, stream>>>(amax, denom, agg, N * HEADS, NF);

        edge_logits <<<gridEdge, tpb, 0, stream>>>(qb, kb, src, dst, alpha, amax, E);
        edge_exp    <<<gridEH,   tpb, 0, stream>>>(alpha, dst, amax, denom, E * HEADS);
        edge_scatter<<<gridEdge, tpb, 0, stream>>>(alpha, denom, vb, src, dst, agg, E);

        combine_relu<<<gridNF, tpb, 0, stream>>>(agg, out, NF);
    }
}
